// RWKVBlock_36498632081997
// MI455X (gfx1250) — matrix operand order repaired and checked by reading, not compiled, not run on hardware
//
#include <hip/hip_runtime.h>
#include <math.h>

#define C_DIM 1024
#define H_DIM 4096
#define B_DIM 8
#define T_DIM 2048
#define M_DIM (B_DIM * T_DIM)

typedef __attribute__((ext_vector_type(16))) __bf16 v16bf;
typedef __attribute__((ext_vector_type(8)))  float  v8f;
typedef __attribute__((ext_vector_type(4)))  unsigned int v4u;
typedef __attribute__((ext_vector_type(8)))  int v8i;
typedef __attribute__((ext_vector_type(4)))  int v4i;
typedef unsigned short ushort_t;

__device__ __forceinline__ ushort_t f2bf(float f) {
  union { float f; unsigned int u; } x; x.f = f;
  unsigned int r = x.u + 0x7FFFu + ((x.u >> 16) & 1u);   // round-to-nearest-even
  return (ushort_t)(r >> 16);
}
__device__ __forceinline__ float bf2f(ushort_t h) {
  union { unsigned int u; float f; } x; x.u = ((unsigned int)h) << 16;
  return x.f;
}

// ---------------- TDM: async 2D tile load Global -> LDS ----------------
// Builds a Tensor DMA Descriptor (D#) per CDNA5 ISA ch.8 and issues
// tensor_load_to_lds. data_size = 2 bytes (bf16). LDS destination padding:
// 4 DWORDs every 16 DWORDs -> 80-byte row stride (bank-conflict-free frags).
__device__ __forceinline__ void tdm_load_tile(unsigned lds_off, const void* gsrc,
                                              int tensor_d0, int tensor_d1,
                                              int tile_d0, int tile_d1,
                                              int stride_e) {
#if defined(__HIP_DEVICE_COMPILE__)
  unsigned long long ga = (unsigned long long)gsrc;
  v4u g0 = { 1u,                                        // count=1, user mode
             lds_off,                                   // lds_addr
             (unsigned)(ga & 0xFFFFFFFFu),              // global_addr[31:0]
             (unsigned)((ga >> 32) & 0x01FFFFFFu) | (2u << 30) };  // addr[56:32] | type=2
  v8i g1;
  g1[0] = (int)((1u << 16) |            // data_size = 1 -> 2 bytes
                (1u << 20) |            // pad_enable
                (3u << 22) |            // pad_interval: 16 DWORDs (64B)
                (3u << 25));            // pad_amount:   4 DWORDs (16B)
  g1[1] = (int)(((unsigned)tensor_d0 & 0xFFFFu) << 16);                  // tensor_dim0 lo16 @ [63:48]
  g1[2] = (int)((((unsigned)tensor_d0 >> 16) & 0xFFFFu) |
                (((unsigned)tensor_d1 & 0xFFFFu) << 16));                // d0 hi16 | d1 lo16
  g1[3] = (int)((((unsigned)tensor_d1 >> 16) & 0xFFFFu) |
                (((unsigned)tile_d0) << 16));                            // d1 hi16 | tile_dim0
  g1[4] = (int)((unsigned)tile_d1 & 0xFFFFu);                            // tile_dim1 | tile_dim2=0
  g1[5] = (int)(unsigned)stride_e;                                       // tensor_dim0_stride lo32
  g1[6] = 0;                                                             // stride hi | dim1_stride lo
  g1[7] = 0;
  v4i z4 = {0, 0, 0, 0};
#if defined(__clang_major__) && (__clang_major__ >= 23)
  v8i z8 = {0, 0, 0, 0, 0, 0, 0, 0};
  __builtin_amdgcn_tensor_load_to_lds(g0, g1, z4, z4, z8, 0);
#else
  __builtin_amdgcn_tensor_load_to_lds(g0, g1, z4, z4, 0);
#endif
#else
  (void)lds_off; (void)gsrc; (void)tensor_d0; (void)tensor_d1;
  (void)tile_d0; (void)tile_d1; (void)stride_e;
#endif
}

template <int N>
__device__ __forceinline__ void wait_tensorcnt() {
#if defined(__HIP_DEVICE_COMPILE__)
  __builtin_amdgcn_s_wait_tensorcnt(N);   // immediate must be a constant
#endif
}

// ---------------- fp32 -> bf16 conversion (weights) ----------------
__global__ __launch_bounds__(256)
void cvt_bf16_kernel(const float* __restrict__ in, ushort_t* __restrict__ out, int n) {
  int i = blockIdx.x * 256 + threadIdx.x;
  if (i < n) out[i] = f2bf(in[i]);
}

// ---------------- LayerNorm: one block (256 thr) per token ----------------
__global__ __launch_bounds__(256)
void layernorm_kernel(const float* __restrict__ x, const float* __restrict__ g,
                      const float* __restrict__ b, float* __restrict__ y) {
  __shared__ float s_sum[8], s_sq[8];
  const int token = blockIdx.x;
  const float* xp = x + (size_t)token * C_DIM;
  float lsum = 0.f, lsq = 0.f;
  for (int i = threadIdx.x; i < C_DIM; i += 256) { float t = xp[i]; lsum += t; lsq += t * t; }
  const int lane = threadIdx.x & 31, wv = threadIdx.x >> 5;
  for (int off = 16; off > 0; off >>= 1) {
    lsum += __shfl_down(lsum, off, 32);
    lsq  += __shfl_down(lsq,  off, 32);
  }
  if (lane == 0) { s_sum[wv] = lsum; s_sq[wv] = lsq; }
  __syncthreads();
  if (wv == 0) {
    lsum = (lane < 8) ? s_sum[lane] : 0.f;
    lsq  = (lane < 8) ? s_sq[lane]  : 0.f;
    for (int off = 4; off > 0; off >>= 1) {
      lsum += __shfl_down(lsum, off, 32);
      lsq  += __shfl_down(lsq,  off, 32);
    }
    if (lane == 0) { s_sum[0] = lsum; s_sq[0] = lsq; }
  }
  __syncthreads();
  const float mean = s_sum[0] * (1.0f / C_DIM);
  const float var  = s_sq[0]  * (1.0f / C_DIM) - mean * mean;
  const float rstd = rsqrtf(var + 1e-5f);
  float* yp = y + (size_t)token * C_DIM;
  for (int i = threadIdx.x; i < C_DIM; i += 256)
    yp[i] = (xp[i] - mean) * rstd * g[i] + b[i];
}

// ---------------- time-shift mix (attention): 3 outputs bf16 ----------------
__global__ __launch_bounds__(256)
void mix_att_kernel(const float* __restrict__ xa, const float* __restrict__ tmk,
                    const float* __restrict__ tmv, const float* __restrict__ tmr,
                    ushort_t* __restrict__ xk, ushort_t* __restrict__ xv,
                    ushort_t* __restrict__ xr) {
  const size_t idx = (size_t)blockIdx.x * 256 + threadIdx.x;
  const int c = (int)(idx & (C_DIM - 1));
  const int t = (int)((idx >> 10) & (T_DIM - 1));   // C_DIM = 1<<10
  const float a  = xa[idx];
  const float xx = (t == 0) ? 0.0f : xa[idx - C_DIM];
  const float mk = tmk[c], mv = tmv[c], mr = tmr[c];
  xk[idx] = f2bf(a * mk + xx * (1.0f - mk));
  xv[idx] = f2bf(a * mv + xx * (1.0f - mv));
  xr[idx] = f2bf(a * mr + xx * (1.0f - mr));
}

// ---------------- time-shift mix (FFN): 2 outputs bf16 ----------------
__global__ __launch_bounds__(256)
void mix_ffn_kernel(const float* __restrict__ xc, const float* __restrict__ cmk,
                    const float* __restrict__ cmr, ushort_t* __restrict__ xk2,
                    ushort_t* __restrict__ xr2) {
  const size_t idx = (size_t)blockIdx.x * 256 + threadIdx.x;
  const int c = (int)(idx & (C_DIM - 1));
  const int t = (int)((idx >> 10) & (T_DIM - 1));
  const float a  = xc[idx];
  const float xx = (t == 0) ? 0.0f : xc[idx - C_DIM];
  const float mk = cmk[c], mr = cmr[c];
  xk2[idx] = f2bf(a * mk + xx * (1.0f - mk));
  xr2[idx] = f2bf(a * mr + xx * (1.0f - mr));
}

// ---------------- WKV sequential scan: one thread per (b,c) ----------------
__global__ __launch_bounds__(256)
void wkv_kernel(const float* __restrict__ k, const float* __restrict__ v,
                const ushort_t* __restrict__ sr, const float* __restrict__ time_decay,
                const float* __restrict__ time_first, ushort_t* __restrict__ rwkv) {
  const int idx = blockIdx.x * 256 + threadIdx.x;   // [0, B*C)
  const int b = idx / C_DIM;
  const int c = idx - b * C_DIM;
  const float w = -expf(time_decay[c]);
  const float u = time_first[c];
  float aa = 0.f, bb = 0.f, pp = -1e38f;
  const size_t base = (size_t)b * T_DIM * C_DIM + c;
  for (int t = 0; t < T_DIM; ++t) {
    const size_t o = base + (size_t)t * C_DIM;
    if (t + 8 < T_DIM) {   // stream-ahead for the strided k/v walk
      __builtin_prefetch((const void*)(k + o + 8 * C_DIM), 0, 1);
      __builtin_prefetch((const void*)(v + o + 8 * C_DIM), 0, 1);
    }
    const float kk = k[o], vv = v[o];
    const float ww = u + kk;
    const float p  = fmaxf(pp, ww);
    const float e1 = expf(pp - p);
    const float e2 = expf(ww - p);
    const float out = (e1 * aa + e2 * vv) / (e1 * bb + e2);
    rwkv[o] = f2bf(bf2f(sr[o]) * out);
    const float ww2 = pp + w;
    const float p2  = fmaxf(ww2, kk);
    const float e1b = expf(ww2 - p2);
    const float e2b = expf(kk - p2);
    aa = e1b * aa + e2b * vv;
    bb = e1b * bb + e2b;
    pp = p2;
  }
}

// ---------------- WMMA bf16 GEMM:  Y[m,n] = epi( sum_k A[m,k]*W[n,k] ) ----------------
// Block = 256 threads (8 waves), tile 128x128; wave tile 64(M) x 32(N).
// TDM double-buffers the 128x32 A and B tiles into LDS (padded rows, 80B stride);
// waves read ISA-layout fragments from LDS:
//   A 16x32 bf16: lanes 0-15/16-31 both hold M=lane%16; K chunks half*8+{0..7},{16..23}
//   B 32x16 bf16: lane half holds K chunks half*8+{0..7},{16..23}, N=lane%16
//   C/D 16x16 f32: vgpr v -> M = v + 8*half, N = lane%16
enum { EPI_F32 = 0, EPI_SIG_BF16 = 1, EPI_RELUSQ_BF16 = 2, EPI_RESID_F32 = 3, EPI_FINAL = 4 };

struct __align__(16) U4 { unsigned int a, b, c, d; };

#define TILE_K 32
#define ROW_STRIDE_E 40                 // padded LDS row stride in bf16 elems (80 B)
#define TILE_BYTES (128 * 80)           // 10240 B per staged tile

template <int EPI>
__global__ __launch_bounds__(256)
void gemm_wmma_kernel(const ushort_t* __restrict__ A, const ushort_t* __restrict__ W,
                      void* __restrict__ Yv, const float* __restrict__ resid,
                      const ushort_t* __restrict__ smat, int M, int N, int K) {
  __shared__ __align__(16) unsigned char lds[2][2][TILE_BYTES];  // [stage][A=0/B=1]

  const int lane  = threadIdx.x & 31;
  const int wave  = threadIdx.x >> 5;
  const int wr    = wave >> 2;          // 0..1  (M direction)
  const int wc    = wave & 3;           // 0..3  (N direction)
  const int nBn   = N >> 7;
  const int bm    = blockIdx.x / nBn;
  const int bn    = blockIdx.x - bm * nBn;
  const int half  = lane >> 4;
  const int l16   = lane & 15;
  const int blockM = bm * 128;
  const int blockN = bn * 128;

  const unsigned ldsA[2] = { (unsigned)(size_t)&lds[0][0][0],
                             (unsigned)(size_t)&lds[1][0][0] };
  const unsigned ldsB[2] = { (unsigned)(size_t)&lds[0][1][0],
                             (unsigned)(size_t)&lds[1][1][0] };

  const int nStages = K / TILE_K;

  if (wave == 0) {
    tdm_load_tile(ldsA[0], A + (size_t)blockM * K, K, 128, TILE_K, 128, K);
    tdm_load_tile(ldsB[0], W + (size_t)blockN * K, K, 128, TILE_K, 128, K);
  }

  v8f acc[4][2] = {};
  union Frag { v16bf v; U4 q[2]; };

  for (int ks = 0; ks < nStages; ++ks) {
    const int cs = ks & 1;
    if (wave == 0) {
      if (ks + 1 < nStages) {
        const int nsb = (ks + 1) & 1;
        const size_t koff = (size_t)(ks + 1) * TILE_K;
        tdm_load_tile(ldsA[nsb], A + (size_t)blockM * K + koff, K, 128, TILE_K, 128, K);
        tdm_load_tile(ldsB[nsb], W + (size_t)blockN * K + koff, K, 128, TILE_K, 128, K);
        wait_tensorcnt<2>();   // in-order completion: stage ks fully in LDS
      } else {
        wait_tensorcnt<0>();
      }
    }
    __syncthreads();         // publish stage ks to all waves

    const ushort_t* lA = (const ushort_t*)&lds[cs][0][0];
    const ushort_t* lB = (const ushort_t*)&lds[cs][1][0];

    Frag a[4], b[2];
#pragma unroll
    for (int im = 0; im < 4; ++im) {
      const int e0 = (wr * 64 + im * 16 + l16) * ROW_STRIDE_E + half * 8;
      a[im].q[0] = *(const U4*)(lA + e0);        // K = half*8 + {0..7}
      a[im].q[1] = *(const U4*)(lA + e0 + 16);   // K = half*8 + {16..23}
    }
#pragma unroll
    for (int in = 0; in < 2; ++in) {
      const int e0 = (wc * 32 + in * 16 + l16) * ROW_STRIDE_E + half * 8;
      b[in].q[0] = *(const U4*)(lB + e0);        // K = half*8 + {0..7}
      b[in].q[1] = *(const U4*)(lB + e0 + 16);   // K = half*8 + {16..23}
    }
#pragma unroll
    for (int im = 0; im < 4; ++im)
#pragma unroll
      for (int in = 0; in < 2; ++in)
        acc[im][in] = __builtin_amdgcn_wmma_f32_16x16x32_bf16(
            false, a[im].v, false, b[in].v, (short)0, acc[im][in], false, false);

    __syncthreads();         // all reads of buffer cs done before TDM overwrites it
  }

  float*    Yf = (float*)Yv;
  ushort_t* Yb = (ushort_t*)Yv;
#pragma unroll
  for (int im = 0; im < 4; ++im)
#pragma unroll
    for (int in = 0; in < 2; ++in)
#pragma unroll
      for (int e = 0; e < 8; ++e) {
        const int row = blockM + wr * 64 + im * 16 + half * 8 + e;
        const int col = blockN + wc * 32 + in * 16 + l16;
        const size_t o = (size_t)row * N + col;
        const float vacc = acc[im][in][e];
        if constexpr (EPI == EPI_F32) {
          Yf[o] = vacc;
        } else if constexpr (EPI == EPI_SIG_BF16) {
          Yb[o] = f2bf(1.0f / (1.0f + expf(-vacc)));
        } else if constexpr (EPI == EPI_RELUSQ_BF16) {
          const float t = fmaxf(vacc, 0.0f);
          Yb[o] = f2bf(t * t);
        } else if constexpr (EPI == EPI_RESID_F32) {
          Yf[o] = resid[o] + vacc;
        } else {  // EPI_FINAL: d_out += sigmoid(r2) * acc
          Yf[o] = Yf[o] + bf2f(smat[o]) * vacc;
        }
      }
}

// ---------------- host launch ----------------
extern "C" void kernel_launch(void* const* d_in, const int* in_sizes, int n_in,
                              void* d_out, int out_size, void* d_ws, size_t ws_size,
                              hipStream_t stream) {
  (void)in_sizes; (void)n_in; (void)out_size; (void)ws_size;

  const float* x      = (const float*)d_in[0];
  const float* ln1_g  = (const float*)d_in[1];
  const float* ln1_b  = (const float*)d_in[2];
  const float* ln2_g  = (const float*)d_in[3];
  const float* ln2_b  = (const float*)d_in[4];
  const float* tdecay = (const float*)d_in[5];
  const float* tfirst = (const float*)d_in[6];
  const float* tmk    = (const float*)d_in[7];
  const float* tmv    = (const float*)d_in[8];
  const float* tmr    = (const float*)d_in[9];
  const float* att_Wk = (const float*)d_in[10];
  const float* att_Wv = (const float*)d_in[11];
  const float* att_Wr = (const float*)d_in[12];
  const float* att_Wo = (const float*)d_in[13];
  const float* cm_k   = (const float*)d_in[14];
  const float* cm_r   = (const float*)d_in[15];
  const float* ffn_Wk = (const float*)d_in[16];
  const float* ffn_Wr = (const float*)d_in[17];
  const float* ffn_Wv = (const float*)d_in[18];

  const int M = M_DIM, Cc = C_DIM, H = H_DIM;
  const size_t CC = (size_t)Cc * Cc, HC = (size_t)H * Cc, MC = (size_t)M * Cc;

  char* ws = (char*)d_ws;
  size_t off = 0;
  auto carve = [&](size_t bytes) -> void* {
    void* p = ws + off;
    off += (bytes + 255) & ~(size_t)255;
    return p;
  };

  ushort_t* wAk = (ushort_t*)carve(CC * 2);
  ushort_t* wAv = (ushort_t*)carve(CC * 2);
  ushort_t* wAr = (ushort_t*)carve(CC * 2);
  ushort_t* wAo = (ushort_t*)carve(CC * 2);
  ushort_t* wFk = (ushort_t*)carve(HC * 2);
  ushort_t* wFr = (ushort_t*)carve(CC * 2);
  ushort_t* wFv = (ushort_t*)carve(HC * 2);
  float*    XA  = (float*)carve(MC * 4);      // LN output (reused for LN2)
  ushort_t* XK  = (ushort_t*)carve(MC * 2);   // also xk2
  ushort_t* XV  = (ushort_t*)carve(MC * 2);   // also xr2
  ushort_t* XR  = (ushort_t*)carve(MC * 2);   // also rwkv
  float*    KB  = (float*)carve(MC * 4);      // k; KB..VB contiguous -> kk(bf16, M*H)
  float*    VB  = (float*)carve(MC * 4);      // v
  ushort_t* SR  = (ushort_t*)carve(MC * 2);   // sigmoid(r); also sigmoid(r2)

  ushort_t* RWKV = XR;
  ushort_t* XK2  = XK;
  ushort_t* XR2  = XV;
  ushort_t* KK   = (ushort_t*)KB;             // M x H bf16 overlays KB+VB
  ushort_t* S2   = SR;

  // --- weights -> bf16 ---
  cvt_bf16_kernel<<<(int)((CC + 255) / 256), 256, 0, stream>>>(att_Wk, wAk, (int)CC);
  cvt_bf16_kernel<<<(int)((CC + 255) / 256), 256, 0, stream>>>(att_Wv, wAv, (int)CC);
  cvt_bf16_kernel<<<(int)((CC + 255) / 256), 256, 0, stream>>>(att_Wr, wAr, (int)CC);
  cvt_bf16_kernel<<<(int)((CC + 255) / 256), 256, 0, stream>>>(att_Wo, wAo, (int)CC);
  cvt_bf16_kernel<<<(int)((HC + 255) / 256), 256, 0, stream>>>(ffn_Wk, wFk, (int)HC);
  cvt_bf16_kernel<<<(int)((CC + 255) / 256), 256, 0, stream>>>(ffn_Wr, wFr, (int)CC);
  cvt_bf16_kernel<<<(int)((HC + 255) / 256), 256, 0, stream>>>(ffn_Wv, wFv, (int)HC);

  // --- attention half ---
  layernorm_kernel<<<M, 256, 0, stream>>>(x, ln1_g, ln1_b, XA);
  mix_att_kernel<<<(int)(MC / 256), 256, 0, stream>>>(XA, tmk, tmv, tmr, XK, XV, XR);

  const dim3 gCC((M / 128) * (Cc / 128));
  const dim3 gCH((M / 128) * (H / 128));
  gemm_wmma_kernel<EPI_F32><<<gCC, 256, 0, stream>>>(XK, wAk, KB, nullptr, nullptr, M, Cc, Cc);
  gemm_wmma_kernel<EPI_F32><<<gCC, 256, 0, stream>>>(XV, wAv, VB, nullptr, nullptr, M, Cc, Cc);
  gemm_wmma_kernel<EPI_SIG_BF16><<<gCC, 256, 0, stream>>>(XR, wAr, SR, nullptr, nullptr, M, Cc, Cc);

  wkv_kernel<<<(B_DIM * C_DIM) / 256, 256, 0, stream>>>(KB, VB, SR, tdecay, tfirst, RWKV);

  // d_out = x + rwkv @ Wo^T
  gemm_wmma_kernel<EPI_RESID_F32><<<gCC, 256, 0, stream>>>(RWKV, wAo, d_out, x, nullptr, M, Cc, Cc);

  // --- channel-mix half ---
  layernorm_kernel<<<M, 256, 0, stream>>>((const float*)d_out, ln2_g, ln2_b, XA);
  mix_ffn_kernel<<<(int)(MC / 256), 256, 0, stream>>>(XA, cm_k, cm_r, XK2, XR2);

  gemm_wmma_kernel<EPI_RELUSQ_BF16><<<gCH, 256, 0, stream>>>(XK2, wFk, KK, nullptr, nullptr, M, H, Cc);
  gemm_wmma_kernel<EPI_SIG_BF16><<<gCC, 256, 0, stream>>>(XR2, wFr, S2, nullptr, nullptr, M, Cc, Cc);
  // d_out += sigmoid(r2) * (kk @ Wv^T)
  gemm_wmma_kernel<EPI_FINAL><<<gCC, 256, 0, stream>>>(KK, wFv, d_out, nullptr, S2, M, Cc, H);
}
